// TanhFixedPointLayer_80788334837949
// MI455X (gfx1250) — matrix operand order repaired and checked by reading, not compiled, not run on hardware
//
#include <hip/hip_runtime.h>

// ---------------------------------------------------------------------------
// TanhFixedPointLayer for MI455X (gfx1250, wave32).
// z_{t+1} = tanh(z @ W^T + x), per-row independent fixed-point iteration.
// Split-bf16 (hi+lo) WMMA GEMM with fp32 accumulation; z resident in 256 KB
// of CDNA5 LDS; W pre-packed into B-fragment order in d_ws and streamed from
// the 192 MB L2. 1024 threads/WG = 32 waves (4 m-tiles x 8 n-groups),
// 8 C-tiles per wave -> 64 accumulator VGPRs (stay under 256, no vgpr_msb).
// ---------------------------------------------------------------------------

typedef __attribute__((ext_vector_type(16))) __bf16 v16bf;
typedef __attribute__((ext_vector_type(8)))  float  v8f;

constexpr int kD      = 1024;          // feature dim
constexpr int kBatch  = 16384;         // rows
constexpr int kMTile  = 64;            // rows per workgroup
constexpr int kIters  = 50;            // MAX_ITER
constexpr int kKSteps = kD / 32;       // 32 WMMA K-steps
constexpr int kNTiles = kD / 16;       // 64 N-tiles
constexpr int kWPackDwords = kNTiles * kKSteps * 32 * 8;  // 524288 dwords/stream

__device__ __forceinline__ unsigned short f2bf(float f) {
  unsigned u = __float_as_uint(f);
  unsigned r = u + 0x7FFFu + ((u >> 16) & 1u);   // round-to-nearest-even
  return (unsigned short)(r >> 16);
}
__device__ __forceinline__ float bf2f(unsigned short h) {
  return __uint_as_float(((unsigned)h) << 16);
}

union FragB { uint4 q[2]; v16bf v; };

// ---------------------------------------------------------------------------
// Prep: split W (fp32 [n][k] row-major) into bf16 hi/lo and scatter into
// B-fragment lane order:  dword index = ((n_tile*32 + ks)*32 + lane)*8 + v
// lane = 16*h + n' ; covers k = ks*32 + 8*h + 16*(v>>2) + 2*(v&3) + {0, 1}.
// ---------------------------------------------------------------------------
__global__ void pack_w_kernel(const float* __restrict__ W,
                              unsigned* __restrict__ whi,
                              unsigned* __restrict__ wlo) {
  unsigned i    = blockIdx.x * blockDim.x + threadIdx.x;
  unsigned v    = i & 7u;
  unsigned lane = (i >> 3) & 31u;
  unsigned ks   = (i >> 8) & 31u;
  unsigned nt   = i >> 13;
  unsigned n = nt * 16 + (lane & 15u);
  unsigned h = lane >> 4;
  unsigned k = ks * 32 + 8 * h + 16 * (v >> 2) + 2 * (v & 3u);
  float w0 = W[n * kD + k];
  float w1 = W[n * kD + k + 1];
  unsigned short h0 = f2bf(w0), h1 = f2bf(w1);
  unsigned short l0 = f2bf(w0 - bf2f(h0)), l1 = f2bf(w1 - bf2f(h1));
  whi[i] = (unsigned)h0 | ((unsigned)h1 << 16);
  wlo[i] = (unsigned)l0 | ((unsigned)l1 << 16);
}

// ---------------------------------------------------------------------------
// Main iterative kernel. 1024 threads = 32 waves: 4 m-tiles x 8 n-groups.
// Each wave: 1 m-tile x 8 n-tiles -> 8 fp32 16x16 accumulators (64 VGPRs).
// ---------------------------------------------------------------------------
__global__ void
__launch_bounds__(1024)
tanh_fixed_point_kernel(const float* __restrict__ x,
                        const unsigned* __restrict__ whi,
                        const unsigned* __restrict__ wlo,
                        float* __restrict__ out) {
  extern __shared__ unsigned short smem[];
  unsigned short* zhi = smem;                 // [kMTile][kD] bf16 hi (128 KB)
  unsigned short* zlo = smem + kMTile * kD;   // [kMTile][kD] bf16 lo (128 KB)

  const unsigned tid  = threadIdx.x;
  const unsigned wave = tid >> 5, lane = tid & 31u;
  const unsigned lr = lane & 15u, lh = lane >> 4;
  const unsigned mt = wave & 3u;              // m-tile within WG (0..3)
  const unsigned ng = wave >> 2;              // n-group (0..7), 8 n-tiles each
  const unsigned m_base = blockIdx.x * kMTile;

  // z0 = 0
  {
    uint4* p = (uint4*)smem;
    const unsigned total = (kMTile * kD * 2u * 2u) / 16u;  // 16384 uint4
    for (unsigned i = tid; i < total; i += blockDim.x) p[i] = make_uint4(0u, 0u, 0u, 0u);
  }
  __syncthreads();

  const unsigned arow = mt * 16 + lr;

#pragma unroll 1
  for (int it = 0; it < kIters; ++it) {
    v8f acc[8];
#pragma unroll
    for (int t = 0; t < 8; ++t)
#pragma unroll
      for (int r = 0; r < 8; ++r) acc[t][r] = 0.0f;

#pragma unroll 1
    for (unsigned ks = 0; ks < (unsigned)kKSteps; ++ks) {
      // A fragments (16x32 bf16) from LDS: lane holds row `arow`,
      // k = ks*32 + 8*lh + {0..7} and +16 -> two ds_load_b128 per fragment.
      FragB ahi, alo;
      const unsigned ka = ks * 32 + 8 * lh;
      ahi.q[0] = *(const uint4*)&zhi[arow * kD + ka];
      ahi.q[1] = *(const uint4*)&zhi[arow * kD + ka + 16];
      alo.q[0] = *(const uint4*)&zlo[arow * kD + ka];
      alo.q[1] = *(const uint4*)&zlo[arow * kD + ka + 16];

#pragma unroll
      for (int t = 0; t < 8; ++t) {
        const unsigned nt    = ng * 8 + (unsigned)t;
        const unsigned fbase = ((nt * (unsigned)kKSteps + ks) * 32 + lane) * 8;
        FragB bhi, blo;
        bhi.q[0] = *(const uint4*)&whi[fbase];
        bhi.q[1] = *(const uint4*)&whi[fbase + 4];
        blo.q[0] = *(const uint4*)&wlo[fbase];
        blo.q[1] = *(const uint4*)&wlo[fbase + 4];
        // hi*hi + hi*lo + lo*hi, fp32 accumulate
        acc[t] = __builtin_amdgcn_wmma_f32_16x16x32_bf16(false, ahi.v, false, bhi.v,
                                                         (short)0, acc[t], false, false);
        acc[t] = __builtin_amdgcn_wmma_f32_16x16x32_bf16(false, ahi.v, false, blo.v,
                                                         (short)0, acc[t], false, false);
        acc[t] = __builtin_amdgcn_wmma_f32_16x16x32_bf16(false, alo.v, false, bhi.v,
                                                         (short)0, acc[t], false, false);
      }
      // prefetch next K-step of the B stream (L2-resident): global_prefetch_b8
      if (ks + 1 < (unsigned)kKSteps) {
        const unsigned nf = (((ng * 8) * (unsigned)kKSteps + ks + 1) * 32 + lane) * 8;
        __builtin_prefetch(&whi[nf], 0, 0);
        __builtin_prefetch(&wlo[nf], 0, 0);
      }
    }

    __syncthreads();  // all waves done reading old z -> safe to update in place

    const bool last = (it == kIters - 1);
#pragma unroll
    for (int t = 0; t < 8; ++t) {
      const unsigned n = (ng * 8 + (unsigned)t) * 16 + lr;
#pragma unroll
      for (int r = 0; r < 8; ++r) {
        const unsigned ml = mt * 16 + 8 * lh + (unsigned)r;   // C layout: m = r + 8*h
        const float zf = tanhf(acc[t][r] + x[(m_base + ml) * kD + n]);
        const unsigned short hb = f2bf(zf);
        zhi[ml * kD + n] = hb;
        zlo[ml * kD + n] = f2bf(zf - bf2f(hb));
        if (last) out[(m_base + ml) * kD + n] = zf;
      }
    }
    __syncthreads();
  }
}

// ---------------------------------------------------------------------------
extern "C" void kernel_launch(void* const* d_in, const int* in_sizes, int n_in,
                              void* d_out, int out_size, void* d_ws, size_t ws_size,
                              hipStream_t stream) {
  (void)in_sizes; (void)n_in; (void)out_size; (void)ws_size;
  const float* x = (const float*)d_in[0];      // [16384,1024] fp32
  const float* W = (const float*)d_in[1];      // [1024,1024]  fp32
  float* out = (float*)d_out;                  // [16384,1024] fp32

  unsigned* whi = (unsigned*)d_ws;             // 2 MB packed bf16-hi fragments
  unsigned* wlo = whi + kWPackDwords;          // 2 MB packed bf16-lo fragments

  pack_w_kernel<<<kWPackDwords / 256, 256, 0, stream>>>(W, whi, wlo);

  const int lds_bytes = kMTile * kD * 2 * (int)sizeof(unsigned short) * 2; // 512 KB
  hipFuncSetAttribute(reinterpret_cast<const void*>(tanh_fixed_point_kernel),
                      hipFuncAttributeMaxDynamicSharedMemorySize, lds_bytes);

  tanh_fixed_point_kernel<<<kBatch / kMTile, 1024, lds_bytes, stream>>>(x, whi, wlo, out);
}
